// MultiHeadMaskedSelfAttention_57492432224475
// MI455X (gfx1250) — matrix operand order repaired and checked by reading, not compiled, not run on hardware
//
#include <hip/hip_runtime.h>

// Problem dims (match reference)
#define BB    4
#define SS    2048
#define DD    1024
#define HH    16
#define DEPTH 64
#define MM    (BB * SS)   // 8192 rows
#define QW    4           // q-tile waves per attention workgroup

typedef _Float16 v16h __attribute__((ext_vector_type(16)));
typedef _Float16 v8h  __attribute__((ext_vector_type(8)));
typedef _Float16 h2v  __attribute__((ext_vector_type(2)));
typedef float    v8f  __attribute__((ext_vector_type(8)));

// D(16x16 f32) = A(16x32 f16) * B(32x16 f16) + C
#define WMMA_F16(a, b, c) \
    __builtin_amdgcn_wmma_f32_16x16x32_f16(false, (a), false, (b), (short)0, (c), false, false)

// Async global->LDS copy, 16B per lane (CDNA5 ISA).
// VGPR operand 0 = LDS byte offset, operand 1 = 64-bit global address.
#define ASYNC_CP_B128(lds_off_u32, gptr)                                        \
    asm volatile("global_load_async_to_lds_b128 %0, %1, off"                    \
                 :: "v"(lds_off_u32), "v"(gptr) : "memory")
#define WAIT_ASYNC0() asm volatile("s_wait_asynccnt 0x0" ::: "memory")

// ---------------------------------------------------------------------------
// Fragment loaders (wave32; contiguous K order chosen here, not the ISA table order)
// A 16x32 f16 (order used here): lane&15 = row M; lanes<16 hold K 0..15, lanes>=16 hold
//   K 16..31; VGPR j packs (K, K+1).
// B 32x16 f16 (same order as A): lane&15 = col N; lanes<16 hold K 0..15, lanes>=16 hold K 16..31.
// C/D 16x16 f32: VGPR r -> row r (lanes<16) / r+8 (lanes>=16), col = lane&15.
// ---------------------------------------------------------------------------
__device__ __forceinline__ v16h frag_a_row(const _Float16* rowp, int aoff /*0 or 8*/) {
    v16h a;
#pragma unroll
    for (int j = 0; j < 4; ++j) {
        h2v p0 = *(const h2v*)(rowp + 2 * aoff + 2 * j);
        h2v p1 = *(const h2v*)(rowp + 8 + 2 * aoff + 2 * j);
        a[2 * j]     = p0[0]; a[2 * j + 1]     = p0[1];
        a[8 + 2 * j] = p1[0]; a[8 + 2 * j + 1] = p1[1];
    }
    return a;
}

__device__ __forceinline__ v16h frag_b_col(const _Float16* colp, int boff /*0 or 16*/) {
    v16h b;
#pragma unroll
    for (int j = 0; j < 8; ++j) {
        h2v p = *(const h2v*)(colp + boff + 2 * j);
        b[2 * j] = p[0]; b[2 * j + 1] = p[1];
    }
    return b;
}

// ---------------------------------------------------------------------------
// f32 -> f16 conversion
// ---------------------------------------------------------------------------
__global__ void cast_f32_f16(const float* __restrict__ in, _Float16* __restrict__ out, int n) {
    int i = blockIdx.x * blockDim.x + threadIdx.x;
    if (i < n) out[i] = (_Float16)in[i];
}

// ---------------------------------------------------------------------------
// GEMM: out = A[M,K](f16) * W[N,K]^T(f16) + bias[N]
// One wave -> one 16x64 output tile (4 C-frags, 4x A reuse per K step).
// OUTMODE: 0 = f16 [M,N] row-major, 1 = f16 V^T per head [B,H,DEPTH,S],
//          2 = f32 [M,N] row-major.
// ---------------------------------------------------------------------------
template <int OUTMODE>
__global__ __launch_bounds__(256) void gemm16x64(const _Float16* __restrict__ A,
                                                 const _Float16* __restrict__ W,
                                                 const float* __restrict__ bias,
                                                 void* __restrict__ out, int N, int K) {
    const int lane = threadIdx.x & 31;
    const int wave = threadIdx.x >> 5;
    const int tm   = blockIdx.x;               // M/16
    const int tn   = blockIdx.y * 8 + wave;    // N/64 tiles of width 64
    const int r15  = lane & 15;
    const int aoff = (lane < 16) ? 0 : 8;
    const int boff = (lane < 16) ? 0 : 16;

    const _Float16* arow = A + (size_t)(tm * 16 + r15) * K;
    const _Float16* wrow[4];
#pragma unroll
    for (int f = 0; f < 4; ++f)
        wrow[f] = W + (size_t)(tn * 64 + f * 16 + r15) * K;  // W row n == B column n

    v8f acc[4] = {};
#pragma unroll 2
    for (int k0 = 0; k0 < K; k0 += 32) {
        const v16h a = frag_a_row(arow + k0, aoff);
#pragma unroll
        for (int f = 0; f < 4; ++f)
            acc[f] = WMMA_F16(a, frag_b_col(wrow[f] + k0, boff), acc[f]);
    }

    const int row0 = tm * 16 + ((lane < 16) ? 0 : 8);  // 8 consecutive M rows per lane
#pragma unroll
    for (int f = 0; f < 4; ++f) {
        const int   col = tn * 64 + f * 16 + r15;
        const float bv  = bias[col];
        if (OUTMODE == 1) {
            // V^T per head: element (b, h, d, s) at (b*DD + col)*SS + s.
            // Lane's 8 rows are consecutive s -> single 16B store.
            const int b  = row0 >> 11;          // row0 / SS
            const int s0 = row0 & (SS - 1);
            v8h pack;
#pragma unroll
            for (int r = 0; r < 8; ++r) pack[r] = (_Float16)(acc[f][r] + bv);
            *(v8h*)((_Float16*)out + ((size_t)b * DD + col) * SS + s0) = pack;
        } else {
#pragma unroll
            for (int r = 0; r < 8; ++r) {
                const float  v   = acc[f][r] + bv;
                const size_t idx = (size_t)(row0 + r) * N + col;
                if (OUTMODE == 2) ((float*)out)[idx]    = v;
                else              ((_Float16*)out)[idx] = (_Float16)v;
            }
        }
    }
}

// ---------------------------------------------------------------------------
// Flash attention, cooperative: one block = 4 waves = 4 adjacent 16-row Q
// tiles of one (b,h). Each 32-kpos step stages the K tile (32x64 f16) and
// V^T tile (64x32 f16) in LDS once per block via async global->LDS copies;
// all 4 waves build B-fragments from LDS.
// Q,K in [B*S, D] f16; V in [B,H,DEPTH,S] f16.
// ---------------------------------------------------------------------------
__global__ __launch_bounds__(32 * QW) void attn_kernel(const _Float16* __restrict__ Qb,
                                                       const _Float16* __restrict__ Kb,
                                                       const _Float16* __restrict__ Vt,
                                                       _Float16* __restrict__ Ob) {
    __shared__ _Float16 ldsK[32 * 64];        // [kpos][depth]   4 KB
    __shared__ _Float16 ldsV[64 * 32];        // [depth][kpos]   4 KB
    __shared__ _Float16 pl[QW][16 * 32];      // per-wave P staging, 4 KB

    const int tid   = threadIdx.x;
    const int lane  = tid & 31;
    const int wave  = tid >> 5;
    const int bh    = blockIdx.y;             // B*H
    const int bidx  = bh >> 4;
    const int h     = bh & 15;
    const int qt    = blockIdx.x * QW + wave;
    const int q0    = qt * 16;
    const int r15   = lane & 15;
    const int aoff  = (lane < 16) ? 0 : 8;
    const int boff  = (lane < 16) ? 0 : 16;
    const int rhalf = (lane < 16) ? 0 : 8;

    const size_t base  = (size_t)bidx * SS * DD + (size_t)h * DEPTH;     // Q/K head base
    const size_t vbase = ((size_t)bidx * DD + h * DEPTH) * SS;           // V^T head base

    // Q tile: 16 x 64 as two A fragments (depth 0..31, 32..63)
    const _Float16* qrow = Qb + base + (size_t)(q0 + r15) * DD;
    const v16h qa0 = frag_a_row(qrow, aoff);
    const v16h qa1 = frag_a_row(qrow + 32, aoff);

    float m[8], l[8];
    v8f   o[4] = {};
#pragma unroll
    for (int r = 0; r < 8; ++r) { m[r] = -3.0e38f; l[r] = 0.0f; }

    // All waves run the same block count (barrier uniformity); blocks past a
    // wave's causal range contribute p=0 / corr=1 numerically.
    const int q0max = (blockIdx.x * QW + (QW - 1)) * 16;
    const int nkb   = (q0max + 15) / 32 + 1;

    for (int kb = 0; kb < nkb; ++kb) {
        const int kbase = kb * 32;

        __syncthreads();  // previous tile fully consumed by all waves

        // Cooperative async staging: 256 x 16B chunks each for K and V^T.
#pragma unroll
        for (int i = 0; i < 2; ++i) {
            const int c    = tid + 128 * i;
            const int krow = c >> 3;                 // 8 chunks per 128B K row
            const int koff = (c & 7) * 8;            // halves
            const _Float16* g = Kb + base + (size_t)(kbase + krow) * DD + koff;
            ASYNC_CP_B128((uint32_t)(size_t)&ldsK[krow * 64 + koff], g);
        }
#pragma unroll
        for (int i = 0; i < 2; ++i) {
            const int c    = tid + 128 * i;
            const int drow = c >> 2;                 // 4 chunks per 64B V^T row
            const int doff = (c & 3) * 8;            // halves
            const _Float16* g = Vt + vbase + (size_t)drow * SS + kbase + doff;
            ASYNC_CP_B128((uint32_t)(size_t)&ldsV[drow * 32 + doff], g);
        }
        WAIT_ASYNC0();
        __syncthreads();  // staged tile visible to all waves

        // Scores: S = Q(16x64) * K_tile^T(64x16), two 16-col halves from LDS
        const _Float16* kc0 = &ldsK[r15 * 64];
        const _Float16* kc1 = &ldsK[(r15 + 16) * 64];
        v8f s0 = {}, s1 = {};
        s0 = WMMA_F16(qa0, frag_b_col(kc0, boff), s0);
        s0 = WMMA_F16(qa1, frag_b_col(kc0 + 32, boff), s0);
        s1 = WMMA_F16(qa0, frag_b_col(kc1, boff), s1);
        s1 = WMMA_F16(qa1, frag_b_col(kc1 + 32, boff), s1);

        // Scale, causal mask (branchless), per-row running max
        float mn[8];
#pragma unroll
        for (int r = 0; r < 8; ++r) {
            const int qr = q0 + r + rhalf;
            const int c0 = kbase + r15;
            float x0 = (c0 > qr)      ? -1.0e9f : s0[r] * 0.125f;  // 1/sqrt(64)
            float x1 = (c0 + 16 > qr) ? -1.0e9f : s1[r] * 0.125f;
            s0[r] = x0; s1[r] = x1;
            float v = fmaxf(x0, x1);
#pragma unroll
            for (int d = 1; d < 16; d <<= 1) v = fmaxf(v, __shfl_xor(v, d, 32));
            mn[r] = fmaxf(m[r], v);
        }

        // Online softmax update
#pragma unroll
        for (int r = 0; r < 8; ++r) {
            const float corr = __expf(m[r] - mn[r]);
            m[r] = mn[r];
            const float p0 = __expf(s0[r] - mn[r]);
            const float p1 = __expf(s1[r] - mn[r]);
            float rs = p0 + p1;
#pragma unroll
            for (int d = 1; d < 16; d <<= 1) rs += __shfl_xor(rs, d, 32);
            l[r] = l[r] * corr + rs;
#pragma unroll
            for (int f = 0; f < 4; ++f) o[f][r] *= corr;
            s0[r] = p0; s1[r] = p1;
        }

        // P (C-layout f32) -> per-wave LDS staging -> A-fragment.
        // Same-wave LDS ops are in-order (DScnt); no cross-wave sharing here.
#pragma unroll
        for (int r = 0; r < 8; ++r) {
            const int row = r + rhalf;
            pl[wave][row * 32 + r15]      = (_Float16)s0[r];
            pl[wave][row * 32 + r15 + 16] = (_Float16)s1[r];
        }
        const v16h pa = frag_a_row(&pl[wave][r15 * 32], aoff);

        // O += P(16x32) * V_tile(32x64) from LDS V^T (kpos contiguous per lane)
#pragma unroll
        for (int f = 0; f < 4; ++f)
            o[f] = WMMA_F16(pa, frag_b_col(&ldsV[(f * 16 + r15) * 32], boff), o[f]);
    }

    // Normalize and store O tile (f16, [B*S, D] layout)
#pragma unroll
    for (int r = 0; r < 8; ++r) {
        const float inv = 1.0f / l[r];
        const int   row = q0 + r + rhalf;
#pragma unroll
        for (int f = 0; f < 4; ++f) {
            Ob[((size_t)bidx * SS + row) * DD + h * DEPTH + f * 16 + r15] =
                (_Float16)(o[f][r] * inv);
        }
    }
}

// ---------------------------------------------------------------------------
// Host-side launch
// ---------------------------------------------------------------------------
extern "C" void kernel_launch(void* const* d_in, const int* in_sizes, int n_in,
                              void* d_out, int out_size, void* d_ws, size_t ws_size,
                              hipStream_t stream) {
    (void)in_sizes; (void)n_in; (void)out_size; (void)ws_size;
    const float* x  = (const float*)d_in[0];
    // d_in[1] is the causal mask; applied analytically in attn_kernel.
    const float* wq = (const float*)d_in[2];
    const float* bq = (const float*)d_in[3];
    const float* wk = (const float*)d_in[4];
    const float* bk = (const float*)d_in[5];
    const float* wv = (const float*)d_in[6];
    const float* bv = (const float*)d_in[7];
    const float* wo = (const float*)d_in[8];
    const float* bo = (const float*)d_in[9];

    char* ws = (char*)d_ws;
    const size_t xh_bytes = (size_t)MM * DD * sizeof(_Float16);   // 16.8 MB
    const size_t w_bytes  = (size_t)DD * DD * sizeof(_Float16);   //  2.1 MB
    _Float16* xh  = (_Float16*)ws;                                // x_f16; reused as attn output
    _Float16* wqh = (_Float16*)(ws + xh_bytes);
    _Float16* wkh = (_Float16*)(ws + xh_bytes + 1 * w_bytes);
    _Float16* wvh = (_Float16*)(ws + xh_bytes + 2 * w_bytes);
    _Float16* woh = (_Float16*)(ws + xh_bytes + 3 * w_bytes);
    _Float16* Qh  = (_Float16*)(ws + xh_bytes + 4 * w_bytes);
    _Float16* Kh  = Qh + (size_t)MM * DD;
    _Float16* Vt  = Kh + (size_t)MM * DD;   // [B, H, DEPTH, S] f16
    // Total workspace: 16.8 + 8.4 + 3*16.8 = ~75.5 MB

    const int nx = MM * DD, nw = DD * DD;
    cast_f32_f16<<<(nx + 255) / 256, 256, 0, stream>>>(x,  xh,  nx);
    cast_f32_f16<<<(nw + 255) / 256, 256, 0, stream>>>(wq, wqh, nw);
    cast_f32_f16<<<(nw + 255) / 256, 256, 0, stream>>>(wk, wkh, nw);
    cast_f32_f16<<<(nw + 255) / 256, 256, 0, stream>>>(wv, wvh, nw);
    cast_f32_f16<<<(nw + 255) / 256, 256, 0, stream>>>(wo, woh, nw);

    const dim3 gg(MM / 16, DD / (64 * 8));  // (512, 2): 8 waves x 16x64 tiles
    gemm16x64<0><<<gg, 256, 0, stream>>>(xh, wqh, bq, Qh,    DD, DD);
    gemm16x64<0><<<gg, 256, 0, stream>>>(xh, wkh, bk, Kh,    DD, DD);
    gemm16x64<1><<<gg, 256, 0, stream>>>(xh, wvh, bv, Vt,    DD, DD);

    attn_kernel<<<dim3(SS / (16 * QW), BB * HH), 32 * QW, 0, stream>>>(Qh, Kh, Vt, xh);

    gemm16x64<2><<<gg, 256, 0, stream>>>(xh, woh, bo, d_out, DD, DD);
}
